// SelfAttentionLayer_1614907703989
// MI455X (gfx1250) — matrix operand order repaired and checked by reading, not compiled, not run on hardware
//
#include <hip/hip_runtime.h>
#include <hip/hip_bf16.h>
#include <math.h>

// ---------------------------------------------------------------------------
// Types for CDNA5 WMMA (wave32): A/B = 16 bf16 per lane, C/D = 8 f32 per lane
// ---------------------------------------------------------------------------
typedef __attribute__((ext_vector_type(16))) __bf16 v16bf;
typedef __attribute__((ext_vector_type(8)))  float  v8f;

// Pointer types for the gfx1250 async global->LDS B128 builtin (probed
// signature: (v4i __device__*, v4i __shared__*, imm offset, imm cpol))
typedef int v4i __attribute__((vector_size(16)));
typedef __attribute__((address_space(1))) v4i* as1_v4i_ptr;
typedef __attribute__((address_space(3))) v4i* as3_v4i_ptr;

#if __has_builtin(__builtin_amdgcn_global_load_async_to_lds_b128)
#define USE_ASYNC_LDS 1
#else
#define USE_ASYNC_LDS 0
#endif

template<int N>
__device__ __forceinline__ void wait_async() {
#if __has_builtin(__builtin_amdgcn_s_wait_asynccnt)
    __builtin_amdgcn_s_wait_asynccnt((unsigned short)N);
#else
    asm volatile("s_wait_asynccnt %0" :: "i"(N) : "memory");
#endif
}

__device__ inline unsigned short f32_to_bf16(float f) {
    union { float f; unsigned int u; } x; x.f = f;
    unsigned int u = x.u;
    unsigned int r = u + 0x7FFFu + ((u >> 16) & 1u);   // round-to-nearest-even
    return (unsigned short)(r >> 16);
}
__device__ inline float bf16_to_f32(unsigned short h) {
    union { unsigned int u; float f; } x; x.u = ((unsigned int)h) << 16;
    return x.f;
}

// LDS K-stride (elements): 40 -> 80 bytes, multiple of 16B (async B128 align),
// non-power-of-two vs the 64 banks to break conflict patterns.
constexpr int BKP = 40;

// ---- WMMA fragment gathers from LDS (contiguous k = 16*half + s, same order for A and B) --------
__device__ __forceinline__ v16bf frag_a(const unsigned short* rowp, int koffA) {
    union { v16bf v; unsigned int u[8]; } f;
    #pragma unroll
    for (int v = 0; v < 8; ++v) {
        int kk = koffA + (v << 1);
        f.u[v] = *(const unsigned int*)&rowp[kk];
    }
    return f.v;
}
__device__ __forceinline__ v16bf frag_b(const unsigned short* colp) {
    union { v16bf v; unsigned int u[8]; } f;
    #pragma unroll
    for (int v = 0; v < 8; ++v) f.u[v] = *(const unsigned int*)&colp[2 * v];
    return f.v;
}
#define WMMA_BF16(A_, B_, C_) \
    __builtin_amdgcn_wmma_f32_16x16x32_bf16(false, (A_), false, (B_), (short)0, (C_), false, false)

// ---------------------------------------------------------------------------
// fp32 -> bf16 conversion (vectorized 4/thread, n is a multiple of 4)
// ---------------------------------------------------------------------------
__global__ __launch_bounds__(256) void cvt_f32_bf16(
    const float* __restrict__ src, unsigned short* __restrict__ dst, int n4)
{
    int i = blockIdx.x * blockDim.x + threadIdx.x;
    if (i >= n4) return;
    float4 f = *(const float4*)(src + (size_t)i * 4);
    union { unsigned long long q; unsigned short s[4]; } o;
    o.s[0] = f32_to_bf16(f.x);
    o.s[1] = f32_to_bf16(f.y);
    o.s[2] = f32_to_bf16(f.z);
    o.s[3] = f32_to_bf16(f.w);
    *(unsigned long long*)(dst + (size_t)i * 4) = o.q;
}

// ---------------------------------------------------------------------------
// NN GEMM: C = scale * (A[MxK] @ B[KxN]), bf16 in, OUTF32 ? f32 : bf16 out.
// WG tile 128x64, 8 waves (4M x 2N), each wave 32x32 = 4 WMMA accumulators.
// Register-pipelined: global loads of tile k+1 issued before tile-k WMMAs.
// B tile is transposed into ldsB[n][k] on the LDS store (fragment reads then
// hit contiguous bf16 pairs).  All dims are multiples of tile sizes.
// ---------------------------------------------------------------------------
template<bool OUTF32>
__global__ __launch_bounds__(256) void wmma_gemm_nn(
    const unsigned short* __restrict__ A, int lda, long long strideA,
    const unsigned short* __restrict__ B, int ldb, long long strideB,
    void* __restrict__ Cv, int ldc, long long strideC,
    int M, int N, int K, float scale)
{
    constexpr int BM = 128, BN = 64, BK = 32;
    __shared__ unsigned short ldsA[BM * BKP];
    __shared__ unsigned short ldsB[BN * BKP];

    const int z = blockIdx.z;
    A += (long long)z * strideA;
    B += (long long)z * strideB;
    float*          Cf = (float*)Cv          + (long long)z * strideC;
    unsigned short* Cb = (unsigned short*)Cv + (long long)z * strideC;

    const int mtile = blockIdx.x * BM;
    const int ntile = blockIdx.y * BN;

    const int tid  = threadIdx.x;
    const int lane = tid & 31;
    const int wid  = tid >> 5;
    const int m0   = (wid & 3) * 32;
    const int n0   = (wid >> 2) * 32;
    const int half = lane >> 4;
    const int l15  = lane & 15;
    const int koffA = half * 16;
    const int koffB = half * 16;

    // cooperative-load coordinates
    const int arow  = tid >> 2;          // A rows arow, arow+64 ; 8 bf16 each
    const int akcol = (tid & 3) * 8;
    const int bkrow = tid >> 3;          // B k-row 0..31
    const int bncol = (tid & 7) * 8;     // 8 consecutive n

    v8f a00 = {}, a01 = {}, a10 = {}, a11 = {};

    // prologue: fetch tile 0 into registers
    uint4 ra0 = *(const uint4*)(A + (long long)(mtile + arow) * lda + akcol);
    uint4 ra1 = *(const uint4*)(A + (long long)(mtile + arow + 64) * lda + akcol);
    union { uint4 q; unsigned short e[8]; } rb;
    rb.q = *(const uint4*)(B + (long long)bkrow * ldb + ntile + bncol);

    for (int k0 = 0; k0 < K; k0 += BK) {
        // commit registers -> LDS
        {
            unsigned int* dl0 = (unsigned int*)&ldsA[arow * BKP + akcol];
            dl0[0] = ra0.x; dl0[1] = ra0.y; dl0[2] = ra0.z; dl0[3] = ra0.w;
            unsigned int* dl1 = (unsigned int*)&ldsA[(arow + 64) * BKP + akcol];
            dl1[0] = ra1.x; dl1[1] = ra1.y; dl1[2] = ra1.z; dl1[3] = ra1.w;
            #pragma unroll
            for (int j = 0; j < 8; ++j)              // transpose into [n][k]
                ldsB[(bncol + j) * BKP + bkrow] = rb.e[j];
        }
        __syncthreads();

        // issue next-tile global loads (latency hides under the WMMA block)
        const int kn = k0 + BK;
        if (kn < K) {
            ra0  = *(const uint4*)(A + (long long)(mtile + arow) * lda + kn + akcol);
            ra1  = *(const uint4*)(A + (long long)(mtile + arow + 64) * lda + kn + akcol);
            rb.q = *(const uint4*)(B + (long long)(kn + bkrow) * ldb + ntile + bncol);
            if (kn + BK < K)  // deep prefetch -> global_prefetch_b8
                __builtin_prefetch(A + (long long)(mtile + arow) * lda + kn + BK + akcol, 0, 2);
        }

        // fragments + 4 WMMAs (2 A frags x 2 B frags)
        v16bf fa0 = frag_a(&ldsA[(m0 +      l15) * BKP], koffA);
        v16bf fa1 = frag_a(&ldsA[(m0 + 16 + l15) * BKP], koffA);
        v16bf fb0 = frag_b(&ldsB[(n0 +      l15) * BKP + koffB]);
        v16bf fb1 = frag_b(&ldsB[(n0 + 16 + l15) * BKP + koffB]);
        a00 = WMMA_BF16(fa0, fb0, a00);
        a01 = WMMA_BF16(fa0, fb1, a01);
        a10 = WMMA_BF16(fa1, fb0, a10);
        a11 = WMMA_BF16(fa1, fb1, a11);
        __syncthreads();
    }

    // epilogue (f32 C/D VGPR layout -> global)
    const int r0 = mtile + m0 + 8 * half;
    const int c0 = ntile + n0 + l15;
    #pragma unroll
    for (int v = 0; v < 8; ++v) {
        long long ra = (long long)(r0 + v) * ldc;
        long long rbn = (long long)(r0 + 16 + v) * ldc;
        if constexpr (OUTF32) {
            Cf[ra  + c0]      = a00[v] * scale;
            Cf[ra  + c0 + 16] = a01[v] * scale;
            Cf[rbn + c0]      = a10[v] * scale;
            Cf[rbn + c0 + 16] = a11[v] * scale;
        } else {
            Cb[ra  + c0]      = f32_to_bf16(a00[v] * scale);
            Cb[ra  + c0 + 16] = f32_to_bf16(a01[v] * scale);
            Cb[rbn + c0]      = f32_to_bf16(a10[v] * scale);
            Cb[rbn + c0 + 16] = f32_to_bf16(a11[v] * scale);
        }
    }
}

// ---------------------------------------------------------------------------
// NT GEMM: C[bf16] = scale * (A[MxK] @ B[NxK]^T).  Both tiles are straight
// row-major copies into LDS -> perfect fit for GLOBAL_LOAD_ASYNC_TO_LDS_B128
// with double-buffered LDS (ASYNCcnt-tracked, overlaps with WMMA).  Falls
// back to the register pipeline if the async builtin is unavailable.
// ---------------------------------------------------------------------------
__global__ __launch_bounds__(256) void wmma_gemm_nt(
    const unsigned short* __restrict__ A, int lda, long long strideA,
    const unsigned short* __restrict__ B, int ldb, long long strideB,
    unsigned short* __restrict__ C, int ldc, long long strideC,
    int M, int N, int K, float scale)
{
    constexpr int BM = 128, BN = 64, BK = 32;

    const int z = blockIdx.z;
    A += (long long)z * strideA;
    B += (long long)z * strideB;
    C += (long long)z * strideC;

    const int mtile = blockIdx.x * BM;
    const int ntile = blockIdx.y * BN;

    const int tid  = threadIdx.x;
    const int lane = tid & 31;
    const int wid  = tid >> 5;
    const int m0   = (wid & 3) * 32;
    const int n0   = (wid >> 2) * 32;
    const int half = lane >> 4;
    const int l15  = lane & 15;
    const int koffA = half * 16;
    const int koffB = half * 16;

    const int arow  = tid >> 2;          // A rows arow, arow+64
    const int akcol = (tid & 3) * 8;
    const int brow  = tid >> 2;          // B(NT) n-row 0..63
    const int bkcol = (tid & 3) * 8;

    v8f a00 = {}, a01 = {}, a10 = {}, a11 = {};

#if USE_ASYNC_LDS
    __shared__ unsigned short ldsA[2][BM * BKP];
    __shared__ unsigned short ldsB[2][BN * BKP];

    auto issue_tile = [&](int k0, int p) {
        const unsigned short* ga0 = A + (long long)(mtile + arow) * lda + k0 + akcol;
        const unsigned short* ga1 = ga0 + (long long)64 * lda;
        const unsigned short* gb  = B + (long long)(ntile + brow) * ldb + k0 + bkcol;
        __builtin_amdgcn_global_load_async_to_lds_b128(
            (as1_v4i_ptr)ga0, (as3_v4i_ptr)&ldsA[p][arow * BKP + akcol], 0, 0);
        __builtin_amdgcn_global_load_async_to_lds_b128(
            (as1_v4i_ptr)ga1, (as3_v4i_ptr)&ldsA[p][(arow + 64) * BKP + akcol], 0, 0);
        __builtin_amdgcn_global_load_async_to_lds_b128(
            (as1_v4i_ptr)gb,  (as3_v4i_ptr)&ldsB[p][brow * BKP + bkcol], 0, 0);
    };

    issue_tile(0, 0);                    // 3 async ops per thread per tile
    int p = 0;
    for (int k0 = 0; k0 < K; k0 += BK) {
        const int kn = k0 + BK;
        if (kn < K) {
            issue_tile(kn, p ^ 1);       // next tile in flight during compute
            wait_async<3>();             // ...but current tile's 3 ops done
        } else {
            wait_async<0>();
        }
        __syncthreads();                 // all waves' async writes visible

        v16bf fa0 = frag_a((const unsigned short*)&ldsA[p][(m0 +      l15) * BKP], koffA);
        v16bf fa1 = frag_a((const unsigned short*)&ldsA[p][(m0 + 16 + l15) * BKP], koffA);
        v16bf fb0 = frag_b((const unsigned short*)&ldsB[p][(n0 +      l15) * BKP + koffB]);
        v16bf fb1 = frag_b((const unsigned short*)&ldsB[p][(n0 + 16 + l15) * BKP + koffB]);
        a00 = WMMA_BF16(fa0, fb0, a00);
        a01 = WMMA_BF16(fa0, fb1, a01);
        a10 = WMMA_BF16(fa1, fb0, a10);
        a11 = WMMA_BF16(fa1, fb1, a11);
        __syncthreads();                 // buffer p free before reuse
        p ^= 1;
    }
#else
    __shared__ unsigned short ldsA[BM * BKP];
    __shared__ unsigned short ldsB[BN * BKP];

    uint4 ra0 = *(const uint4*)(A + (long long)(mtile + arow) * lda + akcol);
    uint4 ra1 = *(const uint4*)(A + (long long)(mtile + arow + 64) * lda + akcol);
    uint4 rb  = *(const uint4*)(B + (long long)(ntile + brow) * ldb + bkcol);

    for (int k0 = 0; k0 < K; k0 += BK) {
        {
            unsigned int* dl0 = (unsigned int*)&ldsA[arow * BKP + akcol];
            dl0[0] = ra0.x; dl0[1] = ra0.y; dl0[2] = ra0.z; dl0[3] = ra0.w;
            unsigned int* dl1 = (unsigned int*)&ldsA[(arow + 64) * BKP + akcol];
            dl1[0] = ra1.x; dl1[1] = ra1.y; dl1[2] = ra1.z; dl1[3] = ra1.w;
            unsigned int* dlb = (unsigned int*)&ldsB[brow * BKP + bkcol];
            dlb[0] = rb.x; dlb[1] = rb.y; dlb[2] = rb.z; dlb[3] = rb.w;
        }
        __syncthreads();
        const int kn = k0 + BK;
        if (kn < K) {
            ra0 = *(const uint4*)(A + (long long)(mtile + arow) * lda + kn + akcol);
            ra1 = *(const uint4*)(A + (long long)(mtile + arow + 64) * lda + kn + akcol);
            rb  = *(const uint4*)(B + (long long)(ntile + brow) * ldb + kn + bkcol);
        }
        v16bf fa0 = frag_a(&ldsA[(m0 +      l15) * BKP], koffA);
        v16bf fa1 = frag_a(&ldsA[(m0 + 16 + l15) * BKP], koffA);
        v16bf fb0 = frag_b(&ldsB[(n0 +      l15) * BKP + koffB]);
        v16bf fb1 = frag_b(&ldsB[(n0 + 16 + l15) * BKP + koffB]);
        a00 = WMMA_BF16(fa0, fb0, a00);
        a01 = WMMA_BF16(fa0, fb1, a01);
        a10 = WMMA_BF16(fa1, fb0, a10);
        a11 = WMMA_BF16(fa1, fb1, a11);
        __syncthreads();
    }
#endif

    const int r0 = mtile + m0 + 8 * half;
    const int c0 = ntile + n0 + l15;
    #pragma unroll
    for (int v = 0; v < 8; ++v) {
        long long ra = (long long)(r0 + v) * ldc;
        long long rb2 = (long long)(r0 + 16 + v) * ldc;
        C[ra  + c0]      = f32_to_bf16(a00[v] * scale);
        C[ra  + c0 + 16] = f32_to_bf16(a01[v] * scale);
        C[rb2 + c0]      = f32_to_bf16(a10[v] * scale);
        C[rb2 + c0 + 16] = f32_to_bf16(a11[v] * scale);
    }
}

// ---------------------------------------------------------------------------
// Column softmax (reference softmax over axis=1 = query axis).
// One 256-thread WG per (b, j): reads scores[b, :, j] (bf16, stride N),
// softmax in f32, writes TRANSPOSED attnT[b, j, :] so the final einsum is a
// plain row-major NN GEMM.  N == 2048 (8 vals/thread).
// ---------------------------------------------------------------------------
__global__ __launch_bounds__(256) void softmax_col_kernel(
    const unsigned short* __restrict__ scores,
    unsigned short* __restrict__ attnT, int Nn)
{
    const int j = blockIdx.x % Nn;
    const int b = blockIdx.x / Nn;
    const unsigned short* S = scores + (long long)b * Nn * Nn + j;
    unsigned short*       T = attnT  + (long long)b * Nn * Nn + (long long)j * Nn;

    __shared__ float red[256];
    const int tid = threadIdx.x;
    float vals[8];
    float lmax = -3.0e38f;
    #pragma unroll
    for (int t = 0; t < 8; ++t) {
        int i = tid + 256 * t;
        float s = bf16_to_f32(S[(long long)i * Nn]);
        vals[t] = s;
        lmax = fmaxf(lmax, s);
    }
    red[tid] = lmax; __syncthreads();
    for (int off = 128; off > 0; off >>= 1) {
        if (tid < off) red[tid] = fmaxf(red[tid], red[tid + off]);
        __syncthreads();
    }
    const float bmax = red[0]; __syncthreads();

    float lsum = 0.f;
    #pragma unroll
    for (int t = 0; t < 8; ++t) { vals[t] = __expf(vals[t] - bmax); lsum += vals[t]; }
    red[tid] = lsum; __syncthreads();
    for (int off = 128; off > 0; off >>= 1) {
        if (tid < off) red[tid] += red[tid + off];
        __syncthreads();
    }
    const float inv = 1.0f / red[0];
    #pragma unroll
    for (int t = 0; t < 8; ++t) {
        int i = tid + 256 * t;
        T[i] = f32_to_bf16(vals[t] * inv);
    }
}

// ---------------------------------------------------------------------------
// Host-side orchestration
// ---------------------------------------------------------------------------
extern "C" void kernel_launch(void* const* d_in, const int* in_sizes, int n_in,
                              void* d_out, int out_size, void* d_ws, size_t ws_size,
                              hipStream_t stream) {
    (void)in_sizes; (void)n_in; (void)out_size; (void)ws_size;
    constexpr int Bb = 4, Nn = 2048, Dd = 1024, Ee = 1024;
    constexpr long long BND = (long long)Bb * Nn * Dd;   // 8M
    constexpr long long DE  = (long long)Dd * Ee;        // 1M
    constexpr long long BNN = (long long)Bb * Nn * Nn;   // 16M

    const float* x  = (const float*)d_in[0];
    const float* Wq = (const float*)d_in[1];
    const float* Wk = (const float*)d_in[2];
    const float* Wv = (const float*)d_in[3];
    float* out = (float*)d_out;

    char* ws = (char*)d_ws;
    size_t off = 0;
    auto alloc = [&](size_t bytes) -> void* {
        void* p = ws + off;
        off += (bytes + 255) & ~(size_t)255;
        return p;
    };
    unsigned short* xb  = (unsigned short*)alloc(BND * 2);   // 16 MiB
    unsigned short* wqb = (unsigned short*)alloc(DE * 2);
    unsigned short* wkb = (unsigned short*)alloc(DE * 2);
    unsigned short* wvb = (unsigned short*)alloc(DE * 2);
    unsigned short* Qb  = (unsigned short*)alloc(BND * 2);   // 16 MiB
    unsigned short* Kb  = (unsigned short*)alloc(BND * 2);   // contiguous after Qb
    unsigned short* Vb  = (unsigned short*)alloc(BND * 2);
    unsigned short* Sc  = (unsigned short*)alloc(BNN * 2);   // 32 MiB
    unsigned short* AT  = Qb;   // attnT aliases dead Q+K region (exactly BNN*2 B)

    // 1) fp32 -> bf16
    {
        int n4;
        n4 = (int)(BND / 4); cvt_f32_bf16<<<(n4 + 255) / 256, 256, 0, stream>>>(x,  xb,  n4);
        n4 = (int)(DE  / 4); cvt_f32_bf16<<<(n4 + 255) / 256, 256, 0, stream>>>(Wq, wqb, n4);
        n4 = (int)(DE  / 4); cvt_f32_bf16<<<(n4 + 255) / 256, 256, 0, stream>>>(Wk, wkb, n4);
        n4 = (int)(DE  / 4); cvt_f32_bf16<<<(n4 + 255) / 256, 256, 0, stream>>>(Wv, wvb, n4);
    }

    dim3 blk(256);
    // 2) Q/K/V = x @ W   (M=8192, N=1024, K=1024), bf16 out
    {
        dim3 g(Bb * Nn / 128, Ee / 64, 1);
        wmma_gemm_nn<false><<<g, blk, 0, stream>>>(
            xb, Dd, 0, wqb, Ee, 0, Qb, Ee, 0, Bb * Nn, Ee, Dd, 1.0f);
        wmma_gemm_nn<false><<<g, blk, 0, stream>>>(
            xb, Dd, 0, wkb, Ee, 0, Kb, Ee, 0, Bb * Nn, Ee, Dd, 1.0f);
        wmma_gemm_nn<false><<<g, blk, 0, stream>>>(
            xb, Dd, 0, wvb, Ee, 0, Vb, Ee, 0, Bb * Nn, Ee, Dd, 1.0f);
    }
    // 3) scores = (Q @ K^T) / 32, batched over B, bf16 out (async-LDS path)
    {
        dim3 g(Nn / 128, Nn / 64, Bb);
        wmma_gemm_nt<<<g, blk, 0, stream>>>(
            Qb, Ee, (long long)Nn * Ee,
            Kb, Ee, (long long)Nn * Ee,
            Sc, Nn, (long long)Nn * Nn,
            Nn, Nn, Ee, 1.0f / 32.0f);
    }
    // 4) softmax over axis=1 (columns), writes transposed attnT
    softmax_col_kernel<<<Bb * Nn, 256, 0, stream>>>(Sc, AT, Nn);
    // 5) out[b,j,e] = sum_i attnT[b,j,i] * V[b,i,e]  -> f32
    {
        dim3 g(Nn / 128, Ee / 64, Bb);
        wmma_gemm_nn<true><<<g, blk, 0, stream>>>(
            AT, Nn, (long long)Nn * Nn,
            Vb, Ee, (long long)Nn * Ee,
            out, Ee, (long long)Nn * Ee,
            Nn, Ee, Nn, 1.0f);
    }
}
